// CachedCompressedLinear_41515153883568
// MI455X (gfx1250) — matrix operand order repaired and checked by reading, not compiled, not run on hardware
//
#include <hip/hip_runtime.h>

typedef __attribute__((ext_vector_type(16))) _Float16 v16h;
typedef __attribute__((ext_vector_type(8)))  float    v8f;
typedef __attribute__((ext_vector_type(8)))  unsigned v8u;
typedef __attribute__((ext_vector_type(4)))  int      v4i;
typedef __attribute__((ext_vector_type(4)))  float    v4f;

#define IN_F   4096
#define OUT_F  11008
#define BATCH  16

// Block = 256 threads (8 waves). Block b owns output tile columns [16b, 16b+16).
// Wave w within the block owns K-chunk [512w, 512w+512): 16 WMMAs of K=32.
// Partial 16x16 f32 tiles are reduced across the 8 waves with ds_add_f32 in LDS,
// then wave 0 applies scale/bias and stores.
//
// Weight dequant (exact): code c in [0,255]; f16 bits (0x6400|c) == 1024+c,
// then packed f16 add of -1152 yields (c-128), integer-exact in f16.
__global__ __launch_bounds__(256) void qlinear_wmma_kernel(
    const float* __restrict__ x,      // [16, 4096] f32
    const int*   __restrict__ wq,     // [11008, 4096] int32 codes in [0,255]
    const float* __restrict__ scale,  // [1]
    const float* __restrict__ bias,   // [11008]
    float*       __restrict__ out)    // [16, 11008]
{
    __shared__ float redC[BATCH * 16];   // 1 KB: accumulated 16x16 tile

    const int tid   = threadIdx.x;
    const int lane  = tid & 31;
    const int wave  = tid >> 5;          // 0..7 -> K chunk
    const int nlane = lane & 15;         // N column within tile / M row for A
    const int hsel  = lane >> 4;         // lane-half select
    const int n0    = blockIdx.x * 16;   // first output column of this tile

    redC[tid] = 0.0f;
    __syncthreads();

    const int kbase = wave * 512;

    // B (32x16 f16 fragment): lane = column n; lanes 0-15 hold K+0..15,
    // lanes 16-31 hold K+16..31; 16 consecutive codes per lane.
    const int* pB = wq + (size_t)(n0 + nlane) * IN_F + kbase + hsel * 16;

    // A (16x32 f16 fragment): lane&15 = row m; lanes 0-15 hold K 0..15,
    // lanes 16-31 hold K 16..31.
    const float* pA = x + (size_t)nlane * IN_F + kbase + hsel * 16;

    const _Float16 moff = (_Float16)(-1152.0f);
    const v16h hoff = {moff, moff, moff, moff, moff, moff, moff, moff,
                       moff, moff, moff, moff, moff, moff, moff, moff};

    v8f acc = {};

#pragma unroll 4
    for (int it = 0; it < 16; ++it) {
        // ---- weight codes: 16 consecutive int32, nontemporal (180 MB stream) ----
        v4i b0 = __builtin_nontemporal_load((const v4i*)(pB + 0));
        v4i b1 = __builtin_nontemporal_load((const v4i*)(pB + 4));
        v4i b2 = __builtin_nontemporal_load((const v4i*)(pB + 8));
        v4i b3 = __builtin_nontemporal_load((const v4i*)(pB + 12));

        // ---- activations: 16 consecutive f32 (L2-resident) ----
        v4f a0 = *(const v4f*)(pA + 0);
        v4f a1 = *(const v4f*)(pA + 4);
        v4f a2 = *(const v4f*)(pA + 8);
        v4f a3 = *(const v4f*)(pA + 12);

        v16h A;
        A[0]  = (_Float16)a0[0];  A[1]  = (_Float16)a0[1];
        A[2]  = (_Float16)a0[2];  A[3]  = (_Float16)a0[3];
        A[4]  = (_Float16)a1[0];  A[5]  = (_Float16)a1[1];
        A[6]  = (_Float16)a1[2];  A[7]  = (_Float16)a1[3];
        A[8]  = (_Float16)a2[0];  A[9]  = (_Float16)a2[1];
        A[10] = (_Float16)a2[2];  A[11] = (_Float16)a2[3];
        A[12] = (_Float16)a3[0];  A[13] = (_Float16)a3[1];
        A[14] = (_Float16)a3[2];  A[15] = (_Float16)a3[3];

        // ---- exact dequant via f16 magic bias (codes known to be in [0,255]) ----
        v8u bw;
        bw[0] = ((unsigned)(b0[0] | (b0[1] << 16))) | 0x64006400u;
        bw[1] = ((unsigned)(b0[2] | (b0[3] << 16))) | 0x64006400u;
        bw[2] = ((unsigned)(b1[0] | (b1[1] << 16))) | 0x64006400u;
        bw[3] = ((unsigned)(b1[2] | (b1[3] << 16))) | 0x64006400u;
        bw[4] = ((unsigned)(b2[0] | (b2[1] << 16))) | 0x64006400u;
        bw[5] = ((unsigned)(b2[2] | (b2[3] << 16))) | 0x64006400u;
        bw[6] = ((unsigned)(b3[0] | (b3[1] << 16))) | 0x64006400u;
        bw[7] = ((unsigned)(b3[2] | (b3[3] << 16))) | 0x64006400u;

        v16h B = __builtin_bit_cast(v16h, bw) + hoff;   // 8x v_pk_add_f16

        // D = A x B + C, f32 accumulate
        acc = __builtin_amdgcn_wmma_f32_16x16x32_f16(
            /*neg_a=*/false, A, /*neg_b=*/false, B,
            /*c_mod=*/(short)0, acc, /*reuse_a=*/false, /*reuse_b=*/false);

        pB += 32;
        pA += 32;
    }

    // ---- cross-wave K reduction: C/D layout -> row m = hsel*8 + r, col nlane ----
#pragma unroll
    for (int r = 0; r < 8; ++r) {
        atomicAdd(&redC[(hsel * 8 + r) * 16 + nlane], acc[r]);
    }
    __syncthreads();

    // ---- wave 0: scale, bias, store ----
    if (wave == 0) {
        const float s  = scale[0];
        const float bv = bias[n0 + nlane];
#pragma unroll
        for (int r = 0; r < 8; ++r) {
            const int m = hsel * 8 + r;
            out[(size_t)m * OUT_F + n0 + nlane] =
                redC[m * 16 + nlane] * s + bv;
        }
    }
}

extern "C" void kernel_launch(void* const* d_in, const int* in_sizes, int n_in,
                              void* d_out, int out_size, void* d_ws, size_t ws_size,
                              hipStream_t stream) {
    const float* x     = (const float*)d_in[0];
    const int*   wq    = (const int*)d_in[1];
    const float* scale = (const float*)d_in[2];
    const float* bias  = (const float*)d_in[3];
    float*       out   = (float*)d_out;

    dim3 grid(OUT_F / 16);   // 688 tiles
    dim3 block(256);         // 8 waves: 1 tile x 8 K-chunks
    qlinear_wmma_kernel<<<grid, block, 0, stream>>>(x, wq, scale, bias, out);
}
